// MODEL_CIFAR10_42528766165359
// MI455X (gfx1250) — matrix operand order repaired and checked by reading, not compiled, not run on hardware
//
#include <hip/hip_runtime.h>
#include <hip/hip_bf16.h>
#include <math.h>

typedef __attribute__((ext_vector_type(16))) _Float16 v16h;
typedef __attribute__((ext_vector_type(8)))  _Float16 v8h;
typedef __attribute__((ext_vector_type(8)))  float    v8f;
typedef __attribute__((ext_vector_type(4)))  float    f4;
typedef __attribute__((ext_vector_type(4)))  unsigned int u32x4;
typedef __attribute__((ext_vector_type(8)))  int      i32x8;
typedef __attribute__((ext_vector_type(4)))  int      i32x4;

#ifndef __has_builtin
#define __has_builtin(x) 0
#endif
#if __has_builtin(__builtin_amdgcn_tensor_load_to_lds) && \
    __has_builtin(__builtin_amdgcn_s_wait_tensorcnt)
#define USE_TDM 1
#else
#define USE_TDM 0
#endif

#define CIN 256          // channel width of every non-stem conv in this net
#define LBS 272          // LDS row pitch in halves (512B row + 32B pad)

// ---------------------------------------------------------------------------
// Specialized implicit-GEMM conv/deconv/GEMM, Cin=256, via
// V_WMMA_F32_16X16X32_F16 (wave32).  Activations NHWC f32; weights pre-packed
// f16 [Npad][Kpad] rows, K ordered (ky*kw+kx)*256 + c.
//   - outer loop: kernel cells (ky,kx) — no integer div/mod anywhere
//   - per cell: 64x256-half B tile moved into LDS by the Tensor Data Mover
//     (TENSOR_LOAD_TO_LDS + s_wait_tensorcnt; cooperative-copy fallback),
//     then 8 fully-unrolled K-chunks x 4 WMMA read it back via ds_load
//   - each wave owns a 16(M) x 64(N) tile; block = 8 waves = 128 M rows
//   - ldil=2 + pad=k-1-p turns this into ConvTranspose2d(k=4,s=2,p=1)
// ---------------------------------------------------------------------------
__global__ __launch_bounds__(256) void conv_wmma_c256(
    const float* __restrict__ in, const _Float16* __restrict__ wp,
    const float* __restrict__ bias, const float* __restrict__ resid,
    float* __restrict__ out,
    int Hin, int Win, int Ho, int Wo, int Cout,
    int kh, int kw, int stride, int pad, int ldil, int relu_in)
{
  __shared__ alignas(64) _Float16 lb[64 * LBS];   // 34 KB of 320 KB/WGP

  const int tid  = threadIdx.x;
  const int lane = tid & 31;
  const int wave = tid >> 5;
  const long mrow = (long)blockIdx.x * 128 + wave * 16 + (lane & 15); // A row
  const int  nb   = blockIdx.y * 64;
  const int  kb   = (lane & 16) ? 16 : 0;  // per-lane K base (contiguous: k = 16h + s)
  const int  Kpad = kh * kw * CIN;
  const int  Npad = (int)gridDim.y * 64;   // rows in packed weight matrix

  const int ow = (int)(mrow % Wo);
  const int oh = (int)((mrow / Wo) % Ho);
  const int bb = (int)(mrow / ((long)Wo * Ho));

#if !USE_TDM
  // cooperative B-staging coordinates: thread -> (row 0..63, 64-half segment)
  const int sr = tid >> 2;
  const int sq = (tid & 3) * 64;
#endif

  v8f acc[4] = {};
  int ky = 0, kx = 0;
  const int ncell = kh * kw;

  for (int cell = 0; cell < ncell; ++cell) {
    // ---- stage 64x256-half weight tile for this cell into LDS ----
    __syncthreads();
#if USE_TDM
    if (wave == 0) {
      // Tensor DMA descriptor (ISA 8.3/8.4): 2D tile, 64 rows x 256 halves,
      // global row pitch Kpad halves, LDS padded to 272-half pitch
      // (pad_interval = 512B rows, pad_amount = 32B).
      const unsigned long long ga =
          (unsigned long long)(size_t)(wp + (size_t)nb * Kpad + cell * CIN);
      const unsigned int ldsoff = (unsigned int)(size_t)&lb[0];
      u32x4 g0;
      g0[0] = 1u;                                   // count=1, user D#
      g0[1] = ldsoff;                               // lds_addr (bytes)
      g0[2] = (unsigned int)ga;                     // global_addr[95:64]
      g0[3] = (unsigned int)((ga >> 32) & 0x01FFFFFFu) | (2u << 30); // type=2
      i32x8 g1;
      g1[0] = (int)((1u << 16)      // data_size = 2 bytes
                  | (1u << 20)      // pad_enable
                  | (6u << 22)      // pad_interval: 128 DWORDs (512B)
                  | (7u << 25));    // pad_amount: 8 DWORDs (32B)
      g1[1] = (int)(((unsigned)Kpad & 0xFFFFu) << 16);               // dim0 lo
      g1[2] = (int)(((unsigned)Kpad >> 16) |
                    (((unsigned)Npad & 0xFFFFu) << 16));             // dim0 hi | dim1 lo
      g1[3] = (int)(((unsigned)Npad >> 16) | (256u << 16));          // dim1 hi | tile_dim0
      g1[4] = (int)(64u);                                            // tile_dim1 | tile_dim2=0
      g1[5] = (int)(unsigned)Kpad;                                   // dim0_stride lo32
      g1[6] = (int)((((unsigned)Kpad & 0xFFFFu) << 16));             // stride hi=0 | dim1_stride lo16
      g1[7] = (int)((unsigned)Kpad >> 16);                           // dim1_stride hi
      i32x4 gz = {};    // groups 2/3: NULL (2D tensor)
      i32x8 gz8 = {};   // trailing group (unused, 2D tensor)
      __builtin_amdgcn_tensor_load_to_lds(g0, g1, gz, gz, gz8, 0);
      __builtin_amdgcn_s_wait_tensorcnt(0);
    }
#else
    {
      const _Float16* gsrc = wp + (size_t)(nb + sr) * Kpad + cell * CIN + sq;
      _Float16* ldst = lb + sr * LBS + sq;
      __builtin_prefetch(gsrc + CIN, 0, 0);     // prefetch next cell's rows
      #pragma unroll
      for (int j = 0; j < 8; ++j)
        *(v8h*)(ldst + j * 8) = *(const v8h*)(gsrc + j * 8);
    }
#endif
    __syncthreads();

    // ---- per-cell A geometry (uniform math, no divisions) ----
    const int ihn = oh * stride - pad + ky;
    const int iwn = ow * stride - pad + kx;
    bool v; int ih, iw;
    if (ldil == 2) {   // lhs-dilated path (ConvTranspose)
      v  = (ihn >= 0) && (iwn >= 0) && ((ihn & 1) == 0) && ((iwn & 1) == 0);
      ih = ihn >> 1; iw = iwn >> 1;
      v  = v && (ih < Hin) && (iw < Win);
    } else {
      ih = ihn; iw = iwn;
      v  = (ihn >= 0) && (iwn >= 0) && (ihn < Hin) && (iwn < Win);
    }
    if (!v) { ih = 0; iw = 0; }
    const float* abase = in + (((size_t)bb * Hin + ih) * Win + iw) * CIN;

    // ---- 8 K-chunks of 32, fully unrolled: 32 WMMA per cell ----
    #pragma unroll
    for (int kk2 = 0; kk2 < CIN; kk2 += 32) {
      v16h a;
      if (v) {
        const float* p = abase + kk2;
        f4 x0 = *(const f4*)(p + kb);
        f4 x1 = *(const f4*)(p + kb + 4);
        f4 x2 = *(const f4*)(p + kb + 8);
        f4 x3 = *(const f4*)(p + kb + 12);
        if (relu_in) {
          #pragma unroll
          for (int e = 0; e < 4; ++e) {
            x0[e] = fmaxf(x0[e], 0.f); x1[e] = fmaxf(x1[e], 0.f);
            x2[e] = fmaxf(x2[e], 0.f); x3[e] = fmaxf(x3[e], 0.f);
          }
        }
        #pragma unroll
        for (int e = 0; e < 4; ++e) {
          a[e]      = (_Float16)x0[e];
          a[4 + e]  = (_Float16)x1[e];
          a[8 + e]  = (_Float16)x2[e];
          a[12 + e] = (_Float16)x3[e];
        }
      } else {
        #pragma unroll
        for (int e = 0; e < 16; ++e) a[e] = (_Float16)0.f;
      }
      const int koff = kk2 + ((lane >> 4) << 4);   // lanes 16-31: K 16..31
      #pragma unroll
      for (int f = 0; f < 4; ++f) {
        const v16h bf = *(const v16h*)(lb + (f * 16 + (lane & 15)) * LBS + koff);
        acc[f] = __builtin_amdgcn_wmma_f32_16x16x32_f16(
            false, a, false, bf, (short)0, acc[f], false, false);
      }
    }
    if (++kx == kw) { kx = 0; ++ky; }
  }

  // ---- epilogue: bias + residual, NHWC store ----
  const int r8 = (lane >> 4) << 3;   // C/D layout: upper half-wave holds M+8
  #pragma unroll
  for (int f = 0; f < 4; ++f) {
    const int n = nb + f * 16 + (lane & 15);
    if (n < Cout) {
      const float bz = bias ? bias[n] : 0.f;
      #pragma unroll
      for (int r = 0; r < 8; ++r) {
        const long mo = (long)blockIdx.x * 128 + wave * 16 + r + r8;
        float vv = acc[f][r] + bz;
        if (resid) vv += resid[(size_t)mo * Cout + n];
        out[(size_t)mo * Cout + n] = vv;
      }
    }
  }
}

// ---------------------------------------------------------------------------
// Generic (runtime-Cin) WMMA conv — used only for the 3-channel stem conv.
// ---------------------------------------------------------------------------
__global__ __launch_bounds__(256) void conv_wmma_generic(
    const float* __restrict__ in, const _Float16* __restrict__ wp,
    const float* __restrict__ bias, const float* __restrict__ resid,
    float* __restrict__ out,
    int Hin, int Win, int Cin,
    int Ho, int Wo, int Cout,
    int kh, int kw, int stride, int pad, int ldil,
    int Kpad, int relu_in)
{
  const int lane = threadIdx.x & 31;
  const int wave = threadIdx.x >> 5;
  const long mrow = (long)blockIdx.x * 128 + wave * 16 + (lane & 15);
  const int  nb   = blockIdx.y * 64;
  const int  K    = kh * kw * Cin;
  const int  kb   = (lane & 16) ? 16 : 0;

  const int ow = (int)(mrow % Wo);
  const int oh = (int)((mrow / Wo) % Ho);
  const int bb = (int)(mrow / ((long)Wo * Ho));

  v8f acc[4] = {};

  for (int kk = 0; kk < Kpad; kk += 32) {
    v16h a;
    #pragma unroll
    for (int e = 0; e < 16; ++e) {
      const int klocal = kb + e;
      const int k = kk + klocal;
      float val = 0.f;
      if (k < K) {
        const int cell = k / Cin;
        const int c = k - cell * Cin;
        const int kyy = cell / kw, kxx = cell - kyy * kw;
        const int ihn = oh * stride - pad + kyy;
        const int iwn = ow * stride - pad + kxx;
        bool v; int ih, iw;
        if (ldil > 1) {
          v  = (ihn >= 0) && (iwn >= 0) && ((ihn % ldil) == 0) && ((iwn % ldil) == 0);
          ih = ihn / ldil; iw = iwn / ldil;
          v  = v && (ih < Hin) && (iw < Win);
        } else {
          ih = ihn; iw = iwn;
          v  = (ihn >= 0) && (iwn >= 0) && (ihn < Hin) && (iwn < Win);
        }
        if (v) val = in[(((size_t)bb * Hin + ih) * Win + iw) * Cin + c];
      }
      if (relu_in) val = fmaxf(val, 0.f);
      a[e] = (_Float16)val;
    }
    const int koff = kk + ((lane >> 4) << 4);
    #pragma unroll
    for (int f = 0; f < 4; ++f) {
      const int n = nb + f * 16 + (lane & 15);
      const v16h bf = *(const v16h*)(wp + (size_t)n * Kpad + koff);
      acc[f] = __builtin_amdgcn_wmma_f32_16x16x32_f16(
          false, a, false, bf, (short)0, acc[f], false, false);
    }
  }

  const int r8 = (lane >> 4) << 3;
  #pragma unroll
  for (int f = 0; f < 4; ++f) {
    const int n = nb + f * 16 + (lane & 15);
    if (n < Cout) {
      const float bz = bias ? bias[n] : 0.f;
      #pragma unroll
      for (int r = 0; r < 8; ++r) {
        const long mo = (long)blockIdx.x * 128 + wave * 16 + r + r8;
        float vv = acc[f][r] + bz;
        if (resid) vv += resid[(size_t)mo * Cout + n];
        out[(size_t)mo * Cout + n] = vv;
      }
    }
  }
}

// ---------------------------------------------------------------------------
// Weight packing: OIHW (conv) or IOHW+flip (deconv) -> f16 [Npad][Kpad]
// ---------------------------------------------------------------------------
__global__ __launch_bounds__(256) void pack_w_kernel(
    const float* __restrict__ w, _Float16* __restrict__ wp,
    int Cout, int Cin, int kh, int kw, int Kpad, int Npad, int deconv)
{
  const size_t idx = (size_t)blockIdx.x * 256 + threadIdx.x;
  const size_t tot = (size_t)Npad * Kpad;
  if (idx >= tot) return;
  const int n = (int)(idx / Kpad);
  const int k = (int)(idx % Kpad);
  const int K = kh * kw * Cin;
  float val = 0.f;
  if (n < Cout && k < K) {
    const int cell = k / Cin;
    const int c = k - cell * Cin;
    const int ky = cell / kw, kx = cell - ky * kw;
    if (!deconv)
      val = w[(((size_t)n * Cin + c) * kh + ky) * kw + kx];
    else  // w is (Cin, Cout, kh, kw); flip spatially, swap I/O
      val = w[(((size_t)c * Cout + n) * kh + (kh - 1 - ky)) * kw + (kw - 1 - kx)];
  }
  wp[idx] = (_Float16)val;
}

// ---------------------------------------------------------------------------
// Layout shims
// ---------------------------------------------------------------------------
__global__ __launch_bounds__(256) void nchw_to_nhwc(
    const float* __restrict__ in, float* __restrict__ out, int B, int C, int H, int W)
{
  size_t i = (size_t)blockIdx.x * 256 + threadIdx.x;
  const size_t tot = (size_t)B * C * H * W;
  if (i >= tot) return;
  const int w = (int)(i % W); size_t t = i / W;
  const int h = (int)(t % H); t /= H;
  const int c = (int)(t % C);
  const int b = (int)(t / C);
  out[(((size_t)b * H + h) * W + w) * C + c] = in[i];
}

__global__ __launch_bounds__(256) void nhwc_to_nchw(
    const float* __restrict__ in, float* __restrict__ out, int B, int C, int H, int W)
{
  size_t i = (size_t)blockIdx.x * 256 + threadIdx.x;
  const size_t tot = (size_t)B * C * H * W;
  if (i >= tot) return;
  const int c = (int)(i % C); size_t t = i / C;
  const int w = (int)(t % W); t /= W;
  const int h = (int)(t % H);
  const int b = (int)(t / H);
  out[(((size_t)b * C + c) * H + h) * W + w] = in[i];
}

__global__ __launch_bounds__(256) void tanh_nhwc_to_nchw(
    const float* __restrict__ in, float* __restrict__ out, int B, int C, int H, int W)
{
  size_t i = (size_t)blockIdx.x * 256 + threadIdx.x;
  const size_t tot = (size_t)B * C * H * W;
  if (i >= tot) return;
  const int c = (int)(i % C); size_t t = i / C;
  const int w = (int)(t % W); t /= W;
  const int h = (int)(t % H);
  const int b = (int)(t / H);
  out[(((size_t)b * C + c) * H + h) * W + w] = tanhf(in[i]);
}

// ---------------------------------------------------------------------------
// VQ helpers
// ---------------------------------------------------------------------------
__global__ __launch_bounds__(256) void row_norms(
    const float* __restrict__ in, float* __restrict__ out, int M, int C)
{
  const int m = blockIdx.x * 256 + threadIdx.x;
  if (m >= M) return;
  const float* p = in + (size_t)m * C;
  float s = 0.f;
  for (int c = 0; c < C; ++c) { const float v = p[c]; s += v * v; }
  out[m] = s;
}

// one wave32 per query row: argmin over T codes, then gather the code vector
// into NHWC workspace AND the NCHW Z_dec output.
__global__ __launch_bounds__(256) void vq_assign(
    const float* __restrict__ dot, const float* __restrict__ qn,
    const float* __restrict__ tn, const float* __restrict__ embd,
    float* __restrict__ zd_nhwc, float* __restrict__ zdec_nchw,
    int M, int T, int C, int Hl, int Wl)
{
  const int lane = threadIdx.x & 31;
  const int wave = threadIdx.x >> 5;
  const int m = blockIdx.x * 8 + wave;
  if (m >= M) return;
  const float* drow = dot + (size_t)m * T;
  const float q = qn[m];
  float best = 3.4e38f; int bidx = 0;
  for (int t = lane; t < T; t += 32) {
    const float d = q + tn[t] - 2.f * drow[t];
    if (d < best) { best = d; bidx = t; }
  }
  #pragma unroll
  for (int off = 16; off > 0; off >>= 1) {
    const float ob = __shfl_xor(best, off, 32);
    const int   oi = __shfl_xor(bidx, off, 32);
    if (ob < best || (ob == best && oi < bidx)) { best = ob; bidx = oi; }
  }
  const float* e = embd + (size_t)bidx * C;
  const int w = m % Wl;
  const int h = (m / Wl) % Hl;
  const int b = m / (Wl * Hl);
  for (int c = lane; c < C; c += 32) {
    const float v = e[c];
    zd_nhwc[(size_t)m * C + c] = v;
    zdec_nchw[(((size_t)b * C + c) * Hl + h) * Wl + w] = v;
  }
}

// one 256-thread block per codebook entry: argmin over M queries, gather row.
__global__ __launch_bounds__(256) void vq_embd_nearest(
    const float* __restrict__ dot, const float* __restrict__ qn,
    const float* __restrict__ tn, const float* __restrict__ zenc,
    float* __restrict__ out, int M, int T, int C)
{
  __shared__ float sval[256];
  __shared__ int   sidx[256];
  const int t = blockIdx.x;
  const int tid = threadIdx.x;
  const float tv = tn[t];
  float best = 3.4e38f; int bidx = 0;
  for (int m = tid; m < M; m += 256) {
    const float d = tv + qn[m] - 2.f * dot[(size_t)m * T + t];
    if (d < best) { best = d; bidx = m; }
  }
  sval[tid] = best; sidx[tid] = bidx;
  __syncthreads();
  for (int s = 128; s > 0; s >>= 1) {
    if (tid < s) {
      if (sval[tid + s] < sval[tid] ||
          (sval[tid + s] == sval[tid] && sidx[tid + s] < sidx[tid])) {
        sval[tid] = sval[tid + s]; sidx[tid] = sidx[tid + s];
      }
    }
    __syncthreads();
  }
  const int mb = sidx[0];
  for (int c = tid; c < C; c += 256)
    out[(size_t)t * C + c] = zenc[(size_t)mb * C + c];
}

// ---------------------------------------------------------------------------
// Orchestration
// ---------------------------------------------------------------------------
extern "C" void kernel_launch(void* const* d_in, const int* in_sizes, int n_in,
                              void* d_out, int out_size, void* d_ws, size_t ws_size,
                              hipStream_t stream)
{
  (void)in_sizes; (void)n_in; (void)out_size; (void)ws_size;
  const float* x     = (const float*)d_in[0];
  const float* c1_w  = (const float*)d_in[1];
  const float* c1_b  = (const float*)d_in[2];
  const float* c2_w  = (const float*)d_in[3];
  const float* c2_b  = (const float*)d_in[4];
  const float* er1w1 = (const float*)d_in[5];
  const float* er1w2 = (const float*)d_in[6];
  const float* er2w1 = (const float*)d_in[7];
  const float* er2w2 = (const float*)d_in[8];
  const float* embd  = (const float*)d_in[9];
  const float* dr1w1 = (const float*)d_in[10];
  const float* dr1w2 = (const float*)d_in[11];
  const float* dr2w1 = (const float*)d_in[12];
  const float* dr2w2 = (const float*)d_in[13];
  const float* dt1_w = (const float*)d_in[14];
  const float* dt1_b = (const float*)d_in[15];
  const float* dt2_w = (const float*)d_in[16];
  const float* dt2_b = (const float*)d_in[17];

  const int B = 32, H = 128, W = 128, Ch = 256, T = 512, Hl = 32, Wl = 32;
  const int Mq = B * Hl * Wl;  // 32768 latent vectors

  // ---- workspace carve (~390 MB; every buffer fully written before read) ---
  char* ws = (char*)d_ws;
  size_t off = 0;
  auto alloc = [&](size_t bytes) -> char* {
    char* p = ws + off; off += (bytes + 255) & ~(size_t)255; return p;
  };
  float*    XH  = (float*)alloc((size_t)B * H * W * 3 * 4);     // x NHWC
  float*    A1  = (float*)alloc((size_t)B * 64 * 64 * Ch * 4);  // 64x64 act (enc & dec)
  float*    A2  = (float*)alloc((size_t)B * Hl * Wl * Ch * 4);
  float*    A3  = (float*)alloc((size_t)B * Hl * Wl * Ch * 4);
  float*    A4  = (float*)alloc((size_t)B * Hl * Wl * Ch * 4);  // Z_enc NHWC (kept)
  float*    TT  = (float*)alloc((size_t)B * Hl * Wl * Ch * 4);  // res temp
  float*    ZD  = (float*)alloc((size_t)B * Hl * Wl * Ch * 4);  // Z_dec NHWC
  float*    DOT = (float*)alloc((size_t)Mq * T * 4);            // Z_enc . embd^T
  float*    QN  = (float*)alloc((size_t)Mq * 4);
  float*    TN  = (float*)alloc((size_t)T * 4);
  float*    U2  = (float*)alloc((size_t)B * H * W * 3 * 4);     // deconv2 NHWC
  _Float16* WP  = (_Float16*)alloc((size_t)256 * 4096 * 2);     // packed weights (reused)

  float* out_x   = (float*)d_out;                    // X_recon  (32,3,128,128)
  float* out_ze  = out_x  + (size_t)B * 3 * H * W;   // Z_enc_ori(32,256,32,32)
  float* out_zd  = out_ze + (size_t)B * Ch * Hl * Wl;// Z_dec    (32,256,32,32)
  float* out_emb = out_zd + (size_t)B * Ch * Hl * Wl;// Z_enc_for_embd (512,256)

  auto pack = [&](const float* w, int Cout, int Cin, int kh, int kw,
                  int Kpad, int Npad, int dec) {
    const size_t tot = (size_t)Npad * Kpad;
    pack_w_kernel<<<dim3((unsigned)((tot + 255) / 256)), dim3(256), 0, stream>>>(
        w, WP, Cout, Cin, kh, kw, Kpad, Npad, dec);
  };
  // Cin==256 fast path (TDM/LDS-staged); stem conv takes the generic kernel.
  auto conv256 = [&](const float* in, const float* bias, const float* resid,
                     float* outp, int Hin, int Win, int Ho, int Wo, int Cout,
                     int kh, int kw, int stride, int pad, int ldil,
                     int Npad, int relu, long M) {
    conv_wmma_c256<<<dim3((unsigned)(M / 128), (unsigned)(Npad / 64)), dim3(256), 0, stream>>>(
        in, WP, bias, resid, outp, Hin, Win, Ho, Wo, Cout,
        kh, kw, stride, pad, ldil, relu);
  };

  // -------- input layout: NCHW -> NHWC --------
  {
    const size_t tot = (size_t)B * 3 * H * W;
    nchw_to_nhwc<<<dim3((unsigned)((tot + 255) / 256)), 256, 0, stream>>>(x, XH, B, 3, H, W);
  }

  // -------- encoder --------
  pack(c1_w, 256, 3, 4, 4, 64, 256, 0);
  conv_wmma_generic<<<dim3((unsigned)((B * 64 * 64) / 128), 4), 256, 0, stream>>>(
      XH, WP, c1_b, nullptr, A1, 128, 128, 3, 64, 64, 256, 4, 4, 2, 1, 1, 64, 0);

  pack(c2_w, 256, 256, 4, 4, 4096, 256, 0);
  conv256(A1, c2_b, nullptr, A2, 64, 64, 32, 32, 256, 4, 4, 2, 1, 1, 256, 0, (long)Mq);

  pack(er1w1, 256, 256, 3, 3, 2304, 256, 0);
  conv256(A2, nullptr, nullptr, TT, 32, 32, 32, 32, 256, 3, 3, 1, 1, 1, 256, 1, (long)Mq);
  pack(er1w2, 256, 256, 1, 1, 256, 256, 0);
  conv256(TT, nullptr, A2, A3, 32, 32, 32, 32, 256, 1, 1, 1, 0, 1, 256, 1, (long)Mq);

  pack(er2w1, 256, 256, 3, 3, 2304, 256, 0);
  conv256(A3, nullptr, nullptr, TT, 32, 32, 32, 32, 256, 3, 3, 1, 1, 1, 256, 1, (long)Mq);
  pack(er2w2, 256, 256, 1, 1, 256, 256, 0);
  conv256(TT, nullptr, A3, A4, 32, 32, 32, 32, 256, 1, 1, 1, 0, 1, 256, 1, (long)Mq);

  // Z_enc_ori output (NHWC -> NCHW)
  {
    const size_t tot = (size_t)B * Ch * Hl * Wl;
    nhwc_to_nchw<<<dim3((unsigned)((tot + 255) / 256)), 256, 0, stream>>>(A4, out_ze, B, Ch, Hl, Wl);
  }

  // -------- vector quantization --------
  pack(embd, T, Ch, 1, 1, 256, 512, 0);   // codebook as GEMM "weights"
  conv256(A4, nullptr, nullptr, DOT, Mq, 1, Mq, 1, T, 1, 1, 1, 0, 1, 512, 0, (long)Mq);

  row_norms<<<dim3((unsigned)(Mq / 256)), 256, 0, stream>>>(A4, QN, Mq, Ch);
  row_norms<<<dim3(2), 256, 0, stream>>>(embd, TN, T, Ch);

  vq_assign<<<dim3((unsigned)(Mq / 8)), 256, 0, stream>>>(
      DOT, QN, TN, embd, ZD, out_zd, Mq, T, Ch, Hl, Wl);

  // -------- decoder --------
  pack(dr1w1, 256, 256, 3, 3, 2304, 256, 0);
  conv256(ZD, nullptr, nullptr, TT, 32, 32, 32, 32, 256, 3, 3, 1, 1, 1, 256, 1, (long)Mq);
  pack(dr1w2, 256, 256, 1, 1, 256, 256, 0);
  conv256(TT, nullptr, ZD, A2, 32, 32, 32, 32, 256, 1, 1, 1, 0, 1, 256, 1, (long)Mq);

  pack(dr2w1, 256, 256, 3, 3, 2304, 256, 0);
  conv256(A2, nullptr, nullptr, TT, 32, 32, 32, 32, 256, 3, 3, 1, 1, 1, 256, 1, (long)Mq);
  pack(dr2w2, 256, 256, 1, 1, 256, 256, 0);
  conv256(TT, nullptr, A2, A3, 32, 32, 32, 32, 256, 1, 1, 1, 0, 1, 256, 1, (long)Mq);

  // ConvTranspose2d(k=4,s=2,p=1): conv over lhs-dilated input, pad e=k-1-p=2
  pack(dt1_w, 256, 256, 4, 4, 4096, 256, 1);
  conv256(A3, dt1_b, nullptr, A1, 32, 32, 64, 64, 256, 4, 4, 1, 2, 2, 256, 0, (long)B * 64 * 64);

  pack(dt2_w, 3, 256, 4, 4, 4096, 64, 1);
  conv256(A1, dt2_b, nullptr, U2, 64, 64, 128, 128, 3, 4, 4, 1, 2, 2, 64, 0, (long)B * H * W);

  // X_recon = tanh, NHWC -> NCHW
  {
    const size_t tot = (size_t)B * 3 * H * W;
    tanh_nhwc_to_nchw<<<dim3((unsigned)((tot + 255) / 256)), 256, 0, stream>>>(U2, out_x, B, 3, H, W);
  }

  // Z_enc_for_embd: nearest latent vector for each codebook entry
  vq_embd_nearest<<<dim3(512), 256, 0, stream>>>(DOT, QN, TN, A4, out_emb, Mq, T, Ch);
}
